// DMPNNLayer_35064113005088
// MI455X (gfx1250) — matrix operand order repaired and checked by reading, not compiled, not run on hardware
//
#include <hip/hip_runtime.h>
#include <hip/hip_bf16.h>
#include <stdint.h>

// ---------------------------------------------------------------------------
// DMPNN layer for MI455X (gfx1250, wave32):
//   bf16 WMMA (v_wmma_f32_16x16x32_bf16, f32 accum) for all three GEMMs,
//   TDM (tensor_load_to_lds + s_wait_tensorcnt) staging of 8KB weight tiles
//   (double-buffered in the message kernel so the DMA overlaps the WMMAs),
//   f32 atomicAdd scatter with hoisted dst indices, fused GRU gate epilogue.
// ---------------------------------------------------------------------------

typedef __attribute__((ext_vector_type(16))) __bf16   v16bf;
typedef __attribute__((ext_vector_type(8)))  float    v8f;
typedef __attribute__((ext_vector_type(4)))  uint32_t u32x4;
typedef __attribute__((ext_vector_type(8)))  uint32_t u32x8;

#define HDIM   256
#define KSTEPS 8      // 256 / 32
#define TILE_M 64     // edges per workgroup
#define WAVES  4      // 128 threads, wave32

// ---- Tensor Data Mover: 1-D copy of one contiguous 8KB bf16 weight tile ----
// group0: count=1 | lds_addr | global_addr[56:0] | type=2
// group1: data_size=1(2B); tensor_dim0=tile_dim0=4096; tensor_dim1=tile_dim1=1;
//         tensor_dim0_stride=4096. Groups 2/3 omitted (<=2D form).
__device__ __forceinline__ void tdm_load_8kb(uint32_t lds_off, const void* gsrc) {
  uint64_t ga = (uint64_t)(uintptr_t)gsrc;
  u32x4 g0;
  g0[0] = 1u;                                                 // count=1 (valid)
  g0[1] = lds_off;                                            // lds_addr (bytes)
  g0[2] = (uint32_t)ga;                                       // global_addr[31:0]
  g0[3] = (uint32_t)((ga >> 32) & 0x01FFFFFFu) | (2u << 30);  // addr[56:32] | type=2
  u32x8 g1;
  g1[0] = 1u << 16;        // data_size=1 (2 bytes)
  g1[1] = 4096u << 16;     // tensor_dim0[15:0] (bits 63:48)
  g1[2] = 1u << 16;        // tensor_dim0[31:16]=0, tensor_dim1[15:0]=1
  g1[3] = 4096u << 16;     // tensor_dim1[31:16]=0, tile_dim0=4096
  g1[4] = 1u;              // tile_dim1=1, tile_dim2=0
  g1[5] = 4096u;           // tensor_dim0_stride[31:0]
  g1[6] = 0u;
  g1[7] = 0u;
  asm volatile("tensor_load_to_lds %0, %1" :: "s"(g0), "s"(g1) : "memory");
}

// ---- A-fragment (16x32 bf16) in contiguous k order (k = 16*half + s) -------------------
__device__ __forceinline__ v16bf load_a_frag_f32(const float* __restrict__ rp,
                                                 int ks, int half) {
  const float* p0 = rp + ks * 32 + half * 16;
  float4 a0 = *reinterpret_cast<const float4*>(p0);
  float4 a1 = *reinterpret_cast<const float4*>(p0 + 4);
  float4 b0 = *reinterpret_cast<const float4*>(p0 + 8);
  float4 b1 = *reinterpret_cast<const float4*>(p0 + 12);
  v16bf f;
  f[0]  = (__bf16)a0.x; f[1]  = (__bf16)a0.y; f[2]  = (__bf16)a0.z; f[3]  = (__bf16)a0.w;
  f[4]  = (__bf16)a1.x; f[5]  = (__bf16)a1.y; f[6]  = (__bf16)a1.z; f[7]  = (__bf16)a1.w;
  f[8]  = (__bf16)b0.x; f[9]  = (__bf16)b0.y; f[10] = (__bf16)b0.z; f[11] = (__bf16)b0.w;
  f[12] = (__bf16)b1.x; f[13] = (__bf16)b1.y; f[14] = (__bf16)b1.z; f[15] = (__bf16)b1.w;
  return f;
}

// ---- B-fragment (32x16 bf16, B[k][n] = W[n][k]) from an LDS tile -----------
__device__ __forceinline__ v16bf load_b_frag_lds(const __bf16* __restrict__ wt_row,
                                                 int ks, int khalf) {
  return *reinterpret_cast<const v16bf*>(wt_row + ks * 32 + khalf * 16);
}

__device__ __forceinline__ float sigmoidf_(float x) {
  return 1.0f / (1.0f + __expf(-x));
}

// ---------------------------------------------------------------------------
__global__ void dmpnn_zero_kernel(float* __restrict__ p, long long n4) {
  long long i = (long long)blockIdx.x * blockDim.x + threadIdx.x;
  if (i < n4) reinterpret_cast<float4*>(p)[i] = make_float4(0.f, 0.f, 0.f, 0.f);
}

__global__ void dmpnn_cvt_weights(const float* __restrict__ wm,
                                  const float* __restrict__ wih,
                                  const float* __restrict__ whh,
                                  __bf16* __restrict__ wmb,
                                  __bf16* __restrict__ wihb,
                                  __bf16* __restrict__ whhb) {
  int i = blockIdx.x * blockDim.x + threadIdx.x;
  const int N1 = HDIM * HDIM;          // 65536
  const int N2 = 3 * HDIM * HDIM;      // 196608
  if (i < N1)                wmb[i]            = (__bf16)wm[i];
  else if (i < N1 + N2)      wihb[i - N1]      = (__bf16)wih[i - N1];
  else if (i < N1 + 2 * N2)  whhb[i - N1 - N2] = (__bf16)whh[i - N1 - N2];
}

// ---------------------------------------------------------------------------
// message = relu(ef @ Wmsg^T + bm); atomicAdd rows into agg at dst indices.
// Weight tiles double-buffered: TDM for tile ct+1 overlaps WMMAs on tile ct.
__global__ void dmpnn_msg_scatter_kernel(const float*  __restrict__ ef,
                                         const int*    __restrict__ eidx,   // [2,E]
                                         const __bf16* __restrict__ Wmb,    // [256,256]
                                         const float*  __restrict__ bm,     // [256]
                                         float*        __restrict__ agg,    // [E,256]
                                         int E) {
  __shared__ __align__(32) __bf16 Wt[2][16][HDIM];   // double-buffered 8KB tiles

  const int lane  = threadIdx.x & 31;
  const int wv    = threadIdx.x >> 5;
  const int half  = lane >> 4;
  const int n     = lane & 15;
  const int etile = blockIdx.x * TILE_M;
  const bool full = (etile + TILE_M) <= E;
  const uint32_t ldsWt = (uint32_t)(uintptr_t)(&Wt[0][0][0]);

  int arow = etile + wv * 16 + n;
  if (arow >= E) arow = E - 1;
  const float* rp = ef + (size_t)arow * HDIM;

  v16bf afrag[KSTEPS];
#pragma unroll
  for (int ks = 0; ks < KSTEPS; ++ks) afrag[ks] = load_a_frag_f32(rp, ks, half);

  // Hoist the 8 dst indices (ct-invariant) out of the column-tile loop.
  int dsti[8];
#pragma unroll
  for (int r = 0; r < 8; ++r) {
    int e = etile + wv * 16 + half * 8 + r;       // C layout: M = r + 8*half
    dsti[r] = eidx[E + (e < E ? e : E - 1)];
  }

  if (wv == 0) tdm_load_8kb(ldsWt, Wmb);          // prologue: tile 0 -> buf 0

  for (int ct = 0; ct < 16; ++ct) {
    __syncthreads();    // all waves done with the buffer tile ct+1 will overwrite
    if (wv == 0) {
      if (ct + 1 < 16) {
        tdm_load_8kb(ldsWt + (uint32_t)((ct + 1) & 1) * 8192u,
                     Wmb + (size_t)(ct + 1) * 16 * HDIM);
        __builtin_amdgcn_s_wait_tensorcnt(1);     // tile ct landed (in-order)
      } else {
        __builtin_amdgcn_s_wait_tensorcnt(0);
      }
    }
    __syncthreads();                              // staged tile visible

    const __bf16* wt = &Wt[ct & 1][n][0];
    v8f acc = {0.f, 0.f, 0.f, 0.f, 0.f, 0.f, 0.f, 0.f};
#pragma unroll
    for (int ks = 0; ks < KSTEPS; ++ks) {
      v16bf bfrag = load_b_frag_lds(wt, ks, half);
      acc = __builtin_amdgcn_wmma_f32_16x16x32_bf16(
          false, afrag[ks], false, bfrag, (short)0, acc, false, false);
    }

    const int   o    = ct * 16 + n;
    const float bias = bm[o];
    if (full) {
#pragma unroll
      for (int r = 0; r < 8; ++r) {
        float v = fmaxf(acc[r] + bias, 0.f);
        atomicAdd(&agg[(size_t)dsti[r] * HDIM + o], v);
      }
    } else {
#pragma unroll
      for (int r = 0; r < 8; ++r) {
        int e = etile + wv * 16 + half * 8 + r;
        if (e < E) {
          float v = fmaxf(acc[r] + bias, 0.f);
          atomicAdd(&agg[(size_t)dsti[r] * HDIM + o], v);
        }
      }
    }
  }
}

// ---------------------------------------------------------------------------
// Fused GRU: gi = agg @ Wih^T + bih, gh = ef @ Whh^T + bhh, gate math, store.
__global__ void dmpnn_gru_kernel(const float*  __restrict__ ef,
                                 const float*  __restrict__ agg,
                                 const __bf16* __restrict__ wihb,  // [768,256]
                                 const __bf16* __restrict__ whhb,  // [768,256]
                                 const float*  __restrict__ bih,
                                 const float*  __restrict__ bhh,
                                 float*        __restrict__ out,   // [E,256]
                                 int E) {
  __shared__ __align__(32) __bf16 Wi[3][16][HDIM];
  __shared__ __align__(32) __bf16 Wh[3][16][HDIM];

  const int lane  = threadIdx.x & 31;
  const int wv    = threadIdx.x >> 5;
  const int half  = lane >> 4;
  const int n     = lane & 15;
  const int etile = blockIdx.x * TILE_M;
  const bool full = (etile + TILE_M) <= E;
  const uint32_t ldsWi = (uint32_t)(uintptr_t)(&Wi[0][0][0]);
  const uint32_t ldsWh = (uint32_t)(uintptr_t)(&Wh[0][0][0]);

  int arow = etile + wv * 16 + n;
  if (arow >= E) arow = E - 1;
  const float* rpx = ef  + (size_t)arow * HDIM;
  const float* rpg = agg + (size_t)arow * HDIM;

  v16bf ax[KSTEPS], ag[KSTEPS];
#pragma unroll
  for (int ks = 0; ks < KSTEPS; ++ks) {
    ax[ks] = load_a_frag_f32(rpx, ks, half);
    ag[ks] = load_a_frag_f32(rpg, ks, half);
  }

  for (int ct = 0; ct < 16; ++ct) {
    __syncthreads();
    if (wv == 0) {
#pragma unroll
      for (int g = 0; g < 3; ++g) {
        tdm_load_8kb(ldsWi + (uint32_t)g * 8192u,
                     wihb + (size_t)(g * HDIM + ct * 16) * HDIM);
        tdm_load_8kb(ldsWh + (uint32_t)g * 8192u,
                     whhb + (size_t)(g * HDIM + ct * 16) * HDIM);
      }
      __builtin_amdgcn_s_wait_tensorcnt(0);
    }
    __syncthreads();

    v8f ir = {0.f,0.f,0.f,0.f,0.f,0.f,0.f,0.f};
    v8f iz = ir, in_ = ir, hr = ir, hz = ir, hn = ir;
#pragma unroll
    for (int ks = 0; ks < KSTEPS; ++ks) {
      v16bf bi0 = load_b_frag_lds(&Wi[0][n][0], ks, half);
      v16bf bi1 = load_b_frag_lds(&Wi[1][n][0], ks, half);
      v16bf bi2 = load_b_frag_lds(&Wi[2][n][0], ks, half);
      v16bf bh0 = load_b_frag_lds(&Wh[0][n][0], ks, half);
      v16bf bh1 = load_b_frag_lds(&Wh[1][n][0], ks, half);
      v16bf bh2 = load_b_frag_lds(&Wh[2][n][0], ks, half);
      ir  = __builtin_amdgcn_wmma_f32_16x16x32_bf16(false, ag[ks], false, bi0, (short)0, ir,  false, false);
      iz  = __builtin_amdgcn_wmma_f32_16x16x32_bf16(false, ag[ks], false, bi1, (short)0, iz,  false, false);
      in_ = __builtin_amdgcn_wmma_f32_16x16x32_bf16(false, ag[ks], false, bi2, (short)0, in_, false, false);
      hr  = __builtin_amdgcn_wmma_f32_16x16x32_bf16(false, ax[ks], false, bh0, (short)0, hr,  false, false);
      hz  = __builtin_amdgcn_wmma_f32_16x16x32_bf16(false, ax[ks], false, bh1, (short)0, hz,  false, false);
      hn  = __builtin_amdgcn_wmma_f32_16x16x32_bf16(false, ax[ks], false, bh2, (short)0, hn,  false, false);
    }

    const int o = ct * 16 + n;
    const float bir = bih[o],           bhr = bhh[o];
    const float biz = bih[HDIM + o],    bhz = bhh[HDIM + o];
    const float bin = bih[2*HDIM + o],  bhn = bhh[2*HDIM + o];
#pragma unroll
    for (int r = 0; r < 8; ++r) {
      int e = etile + wv * 16 + half * 8 + r;     // C layout: M = r + 8*half
      if (full || e < E) {
        float xv = ef[(size_t)e * HDIM + o];
        float rr = sigmoidf_((ir[r] + bir) + (hr[r] + bhr));
        float zz = sigmoidf_((iz[r] + biz) + (hz[r] + bhz));
        float nn = tanhf((in_[r] + bin) + rr * (hn[r] + bhn));
        out[(size_t)e * HDIM + o] = (1.0f - zz) * nn + zz * xv;
      }
    }
  }
}

// ---------------------------------------------------------------------------
extern "C" void kernel_launch(void* const* d_in, const int* in_sizes, int n_in,
                              void* d_out, int out_size, void* d_ws, size_t ws_size,
                              hipStream_t stream) {
  const float* ef   = (const float*)d_in[0];
  const int*   eidx = (const int*)  d_in[1];
  const float* Wm   = (const float*)d_in[2];
  const float* bm   = (const float*)d_in[3];
  const float* wih  = (const float*)d_in[4];
  const float* whh  = (const float*)d_in[5];
  const float* bih  = (const float*)d_in[6];
  const float* bhh  = (const float*)d_in[7];
  float*       out  = (float*)d_out;

  const int E = in_sizes[0] / HDIM;

  // Workspace: agg f32 [E,256] | Wmsg bf16 | Wih bf16 | Whh bf16
  float* agg = (float*)d_ws;
  size_t off = ((size_t)E * HDIM * sizeof(float) + 255) & ~(size_t)255;
  __bf16* wmb  = (__bf16*)((char*)d_ws + off);
  __bf16* wihb = wmb  + HDIM * HDIM;
  __bf16* whhb = wihb + 3 * HDIM * HDIM;

  long long n4 = (long long)E * HDIM / 4;
  dmpnn_zero_kernel<<<(unsigned)((n4 + 255) / 256), 256, 0, stream>>>(agg, n4);

  const int wtot = 7 * HDIM * HDIM;
  dmpnn_cvt_weights<<<(wtot + 255) / 256, 256, 0, stream>>>(Wm, wih, whh, wmb, wihb, whhb);

  const unsigned nblk = (unsigned)((E + TILE_M - 1) / TILE_M);
  dmpnn_msg_scatter_kernel<<<nblk, WAVES * 32, 0, stream>>>(ef, eidx, wmb, bm, agg, E);
  dmpnn_gru_kernel<<<nblk, WAVES * 32, 0, stream>>>(ef, agg, wihb, whhb, bih, bhh, out, E);
}
